// GAT_50362786513306
// MI455X (gfx1250) — matrix operand order repaired and checked by reading, not compiled, not run on hardware
//
#include <hip/hip_runtime.h>
#include <math.h>

#define Hh 4
#define D_ 32
#define HD 128   // Hh * D_

typedef __attribute__((ext_vector_type(16))) _Float16 v16h;
typedef __attribute__((ext_vector_type(8)))  _Float16 v8h;
typedef __attribute__((ext_vector_type(8)))  float    v8f;

// ---- monotonic float <-> uint encoding for atomic max --------------------
__device__ __forceinline__ unsigned f32_enc(float x) {
  unsigned u = __float_as_uint(x);
  return (u & 0x80000000u) ? ~u : (u | 0x80000000u);
}
__device__ __forceinline__ float f32_dec(unsigned v) {
  return (v & 0x80000000u) ? __uint_as_float(v & 0x7fffffffu)
                           : __uint_as_float(~v);
}

// ---- f32 -> f16 bulk convert (once per layer; halves GEMM read bytes) ----
__global__ void cvt_f16(const float* __restrict__ A, _Float16* __restrict__ Ah,
                        int total) {
  int i = blockIdx.x * blockDim.x + threadIdx.x;
  if (i >= total) return;
  Ah[i] = (_Float16)A[i];
}

// ---- pack W[K,128] f32 into f16 WMMA B lane layout -----------------------
// Bp layout: [tile_n(8)][kb(K/32)][lane(32)][j(16)]
// lane<16: B rows K=kb*32+(j<8 ? j : 16+j-8)   , col = tile*16 + lane
// lane>=16: B rows K=kb*32+8+(j<8 ? j : 16+j-8), col = tile*16 + lane-16
__global__ void pack_W_f16(const float* __restrict__ W, _Float16* __restrict__ Bp,
                           int K) {
  int nkb = K >> 5;
  int total = nkb * 8 * 32 * 16;
  int i = blockIdx.x * blockDim.x + threadIdx.x;
  if (i >= total) return;
  int j    = i & 15;
  int lane = (i >> 4) & 31;
  int kb   = (i >> 9) % nkb;
  int tile = i / (nkb << 9);
  int k    = kb * 32 + (lane >> 4) * 8 + (j & 7) + (j >> 3) * 16;
  int col  = tile * 16 + (lane & 15);
  Bp[i] = (_Float16)W[(size_t)k * HD + col];
}

// ---- GEMM: C[N,128] = Ah[N,K] @ W[K,128], f16 WMMA, f32 accumulate -------
// blockDim = 256 (8 waves). Wave w computes tile rows [16*bx,16*bx+16),
// cols [16*w, 16*w+16). Inner loop: 4x global_load_b128 + 1x v_wmma.
__global__ void gat_gemm_wmma(const _Float16* __restrict__ Ah,
                              const _Float16* __restrict__ Bp,
                              float* __restrict__ C, int K) {
  const int wave = threadIdx.x >> 5;     // 0..7 -> column tile
  const int lane = threadIdx.x & 31;
  const int half = lane >> 4;            // 0: lanes 0-15, 1: lanes 16-31
  const int l16  = lane & 15;
  const int row  = blockIdx.x * 16 + l16;
  const int col  = wave * 16 + l16;
  const int nkb  = K >> 5;

  // A 16x32 f16 layout (ISA 7.12.2): elems 0-7 = K[half*8 .. +8),
  //                                  elems 8-15 = K[16+half*8 .. +8)
  const _Float16* __restrict__ pa = Ah + (size_t)row * K + half * 8;
  const _Float16* __restrict__ pb = Bp + ((size_t)(wave * nkb) * 32 + lane) * 16;

  v8f acc = {};
  for (int kb = 0; kb < nkb; ++kb) {
    v8h a0 = *(const v8h*)(pa);
    v8h a1 = *(const v8h*)(pa + 16);
    v8h b0 = *(const v8h*)(pb);
    v8h b1 = *(const v8h*)(pb + 8);
    __builtin_prefetch(pa + 32, 0, 3);   // next A k-block -> global_prefetch_b8
    v16h a = __builtin_shufflevector(a0, a1, 0,1,2,3,4,5,6,7,8,9,10,11,12,13,14,15);
    v16h b = __builtin_shufflevector(b0, b1, 0,1,2,3,4,5,6,7,8,9,10,11,12,13,14,15);
    acc = __builtin_amdgcn_wmma_f32_16x16x32_f16(
        /*neg_a=*/false, a, /*neg_b=*/false, b,
        /*c_mod=*/(short)0, acc, /*reuse_a=*/false, /*reuse_b=*/false);
    pa += 32;
    pb += 32 * 16;
  }
  // C/D layout: VGPR i -> M = i + 8*half, N = l16
#pragma unroll
  for (int i = 0; i < 8; ++i) {
    C[(size_t)(blockIdx.x * 16 + half * 8 + i) * HD + col] = acc[i];
  }
}

// ---- per-(node,head) attention scores: el = <h, al>, er = <h, ar> --------
__global__ void attn_scores(const float* __restrict__ h,
                            const float* __restrict__ al,
                            const float* __restrict__ ar,
                            float* __restrict__ el, float* __restrict__ er,
                            int NH) {
  int t    = blockIdx.x * blockDim.x + threadIdx.x;
  int wid  = t >> 5;
  int lane = t & 31;
  if (wid >= NH) return;
  int hh = wid & (Hh - 1);
  float v  = h[(size_t)wid * D_ + lane];   // h flat [N][H][D]; wid = n*H+hh
  float sl = v * al[hh * D_ + lane];
  float sr = v * ar[hh * D_ + lane];
#pragma unroll
  for (int off = 16; off > 0; off >>= 1) {
    sl += __shfl_xor(sl, off, 32);
    sr += __shfl_xor(sr, off, 32);
  }
  if (lane == 0) { el[wid] = sl; er[wid] = sr; }
}

// ---- edge pass 1: segment max of leaky_relu(el[src]+er[dst]) -------------
__global__ void edge_max_k(const int* __restrict__ src, const int* __restrict__ dst,
                           const float* __restrict__ el, const float* __restrict__ er,
                           unsigned* __restrict__ menc, int E) {
  int i = blockIdx.x * blockDim.x + threadIdx.x;
  if (i >= E * Hh) return;
  int e = i >> 2, hh = i & 3;
  float v = el[src[e] * Hh + hh] + er[dst[e] * Hh + hh];
  v = v < 0.f ? 0.2f * v : v;
  atomicMax(&menc[dst[e] * Hh + hh], f32_enc(v));
}

// ---- edge pass 2: ex = exp(e - m[dst]); segment sum ----------------------
__global__ void edge_exp_k(const int* __restrict__ src, const int* __restrict__ dst,
                           const float* __restrict__ el, const float* __restrict__ er,
                           const unsigned* __restrict__ menc,
                           float* __restrict__ exbuf, float* __restrict__ ssum, int E) {
  int i = blockIdx.x * blockDim.x + threadIdx.x;
  if (i >= E * Hh) return;
  int e = i >> 2, hh = i & 3;
  int d = dst[e];
  float v = el[src[e] * Hh + hh] + er[d * Hh + hh];
  v = v < 0.f ? 0.2f * v : v;
  float m  = f32_dec(menc[d * Hh + hh]);
  float ex = __expf(v - m);
  exbuf[i] = ex;
  atomicAdd(&ssum[d * Hh + hh], ex);
}

// ---- edge pass 2b: alpha = ex / s[dst]  (folds the divide once) ----------
__global__ void edge_alpha_k(const int* __restrict__ dst,
                             const float* __restrict__ ssum,
                             float* __restrict__ exbuf, int E) {
  int i = blockIdx.x * blockDim.x + threadIdx.x;
  if (i >= E * Hh) return;
  int e = i >> 2, hh = i & 3;
  exbuf[i] = exbuf[i] / ssum[dst[e] * Hh + hh];
}

// ---- edge pass 3: agg[dst] += alpha * h[src] (1 thread per (e,h,d)) ------
__global__ void edge_agg_k(const int* __restrict__ src, const int* __restrict__ dst,
                           const float* __restrict__ alpha,
                           const float* __restrict__ h, float* __restrict__ agg,
                           int total) {
  int i = blockIdx.x * blockDim.x + threadIdx.x;
  if (i >= total) return;
  int e  = i >> 7;          // edge
  int r  = i & 127;         // h*32 + d
  int hh = r >> 5;
  atomicAdd(&agg[(size_t)dst[e] * HD + r],
            h[(size_t)src[e] * HD + r] * alpha[e * Hh + hh]);
}

// ---- finalize: out = act(agg + bias). mode 0: relu, 1: elu ---------------
__global__ void finalize_act(const float* __restrict__ agg,
                             const float* __restrict__ bias,
                             float* __restrict__ out, int total, int mode) {
  int i = blockIdx.x * blockDim.x + threadIdx.x;
  if (i >= total) return;
  float v = agg[i] + bias[i & (HD - 1)];
  if (mode == 0) v = v > 0.f ? v : 0.f;
  else           v = v > 0.f ? v : (__expf(v) - 1.0f);
  out[i] = v;
}

// ---- finalize layer 3: mean over heads of (agg + bias) -------------------
__global__ void finalize_mean(const float* __restrict__ agg,
                              const float* __restrict__ bias,
                              float* __restrict__ out, int total /* N*32 */) {
  int i = blockIdx.x * blockDim.x + threadIdx.x;
  if (i >= total) return;
  int n = i >> 5, d = i & 31;
  float sum = 0.f;
#pragma unroll
  for (int hh = 0; hh < Hh; ++hh)
    sum += agg[(size_t)n * HD + hh * D_ + d] + bias[hh * D_ + d];
  out[i] = sum * 0.25f;
}

// ---- host-side layer driver ----------------------------------------------
static void run_layer(const float* Ain, int K, const float* W,
                      const float* al, const float* ar,
                      const int* src, const int* dst, int N, int E,
                      _Float16* Ah, _Float16* Bp,
                      float* h, float* agg, float* el, float* er,
                      unsigned* menc, float* ssum, float* exbuf,
                      hipStream_t stream) {
  int totA = N * K;
  cvt_f16<<<(totA + 255) / 256, 256, 0, stream>>>(Ain, Ah, totA);
  int totB = (K >> 5) * 8 * 32 * 16;
  pack_W_f16<<<(totB + 255) / 256, 256, 0, stream>>>(W, Bp, K);

  gat_gemm_wmma<<<N / 16, 256, 0, stream>>>(Ah, Bp, h, K);

  int NH = N * Hh;
  attn_scores<<<(NH * 32 + 255) / 256, 256, 0, stream>>>(h, al, ar, el, er, NH);

  hipMemsetAsync(menc, 0, (size_t)NH * 4, stream);  // 0 < enc(x) for all real x
  hipMemsetAsync(ssum, 0, (size_t)NH * 4, stream);
  hipMemsetAsync(agg, 0, (size_t)N * HD * 4, stream);

  int EH = E * Hh;
  edge_max_k  <<<(EH + 255) / 256, 256, 0, stream>>>(src, dst, el, er, menc, E);
  edge_exp_k  <<<(EH + 255) / 256, 256, 0, stream>>>(src, dst, el, er, menc, exbuf, ssum, E);
  edge_alpha_k<<<(EH + 255) / 256, 256, 0, stream>>>(dst, ssum, exbuf, E);
  int tot = E * HD;
  edge_agg_k  <<<(tot + 255) / 256, 256, 0, stream>>>(src, dst, exbuf, h, agg, tot);
}

extern "C" void kernel_launch(void* const* d_in, const int* in_sizes, int n_in,
                              void* d_out, int out_size, void* d_ws, size_t ws_size,
                              hipStream_t stream) {
  const float* x   = (const float*)d_in[0];
  const int*   src = (const int*)  d_in[1];
  const int*   dst = (const int*)  d_in[2];
  const float* W1  = (const float*)d_in[3];
  const float* al1 = (const float*)d_in[4];
  const float* ar1 = (const float*)d_in[5];
  const float* b1  = (const float*)d_in[6];
  const float* W2  = (const float*)d_in[7];
  const float* al2 = (const float*)d_in[8];
  const float* ar2 = (const float*)d_in[9];
  const float* b2  = (const float*)d_in[10];
  const float* W3  = (const float*)d_in[11];
  const float* al3 = (const float*)d_in[12];
  const float* ar3 = (const float*)d_in[13];
  const float* b3  = (const float*)d_in[14];

  const int N = in_sizes[0] / 256;   // 50000
  const int E = in_sizes[1];         // 800000

  char* ws = (char*)d_ws;
  size_t off = 0;
  auto carve = [&](size_t bytes) {
    void* p = ws + off;
    off += (bytes + 255) & ~(size_t)255;
    return p;
  };
  float*    h     = (float*)   carve((size_t)N * HD * 4);
  float*    agg   = (float*)   carve((size_t)N * HD * 4);
  float*    buf   = (float*)   carve((size_t)N * HD * 4);
  _Float16* Ah    = (_Float16*)carve((size_t)N * 256 * 2);     // max K = 256
  _Float16* Bp    = (_Float16*)carve((size_t)8 * 8 * 32 * 16 * 2);
  float*    el    = (float*)   carve((size_t)N * Hh * 4);
  float*    er    = (float*)   carve((size_t)N * Hh * 4);
  unsigned* menc  = (unsigned*)carve((size_t)N * Hh * 4);
  float*    ssum  = (float*)   carve((size_t)N * Hh * 4);
  float*    exbuf = (float*)   carve((size_t)E * Hh * 4);

  int totNF = N * HD;

  // Layer 1: x[N,256] -> relu -> buf[N,128]
  run_layer(x, 256, W1, al1, ar1, src, dst, N, E, Ah, Bp,
            h, agg, el, er, menc, ssum, exbuf, stream);
  finalize_act<<<(totNF + 255) / 256, 256, 0, stream>>>(agg, b1, buf, totNF, 0);

  // Layer 2: buf[N,128] -> elu -> buf[N,128] (in-place safe: elementwise)
  run_layer(buf, HD, W2, al2, ar2, src, dst, N, E, Ah, Bp,
            h, agg, el, er, menc, ssum, exbuf, stream);
  finalize_act<<<(totNF + 255) / 256, 256, 0, stream>>>(agg, b2, buf, totNF, 1);

  // Layer 3: buf[N,128] -> mean over heads -> d_out[N,32]
  run_layer(buf, HD, W3, al3, ar3, src, dst, N, E, Ah, Bp,
            h, agg, el, er, menc, ssum, exbuf, stream);
  finalize_mean<<<(N * D_ + 255) / 256, 256, 0, stream>>>(agg, b3, (float*)d_out, N * D_);
}
